// EgoActorCritic_10015863734957
// MI455X (gfx1250) — matrix operand order repaired and checked by reading, not compiled, not run on hardware
//
#include <hip/hip_runtime.h>
#include <hip/hip_bf16.h>

#define RDIM 256
#define NDIM 2048
#define KDIM 64
#define CDIM 16
#define DDIM 64
#define HDIM 128
#define NEGV  (-1e9f)
#define EPSV  (1e-5f)
#define PAIRS 8
#define TN    64            // critic nodes per tile
#define NTILES (NDIM / TN)  // 32

typedef __attribute__((ext_vector_type(16))) _Float16 v16h;
typedef __attribute__((ext_vector_type(8)))  _Float16 v8h;
typedef __attribute__((ext_vector_type(8)))  float    v8f;
typedef __attribute__((ext_vector_type(4)))  float    v4f;

// ---- A fragment (16x32 f16) from an f16 LDS tile; base = tile + kbase halves ----
// ISA layout: lanes 0-15 hold M=lane, K = kbase + {0..7, 16..23};
//             lanes 16-31 hold M=lane-16, K = kbase + {8..15, 24..31}.
__device__ __forceinline__ v16h lds_a_frag(const _Float16* base, int stride_h) {
  const int lane = threadIdx.x & 31;
  const int m    = lane & 15;
  const int koff = (lane < 16) ? 0 : 8;
  const _Float16* p = base + m * stride_h + koff;
  v8h lo = *(const v8h*)(p);
  v8h hi = *(const v8h*)(p + 16);
  v16h a;
#pragma unroll
  for (int i = 0; i < 8; ++i) { a[i] = lo[i]; a[i + 8] = hi[i]; }
  return a;
}

// ---- A fragment from an f32 LDS tile (convert on the fly) ----
__device__ __forceinline__ v16h lds_a_frag_f32(const float* base, int stride_f, int kbase) {
  const int lane = threadIdx.x & 31;
  const int m    = lane & 15;
  const int koff = (lane < 16) ? 0 : 8;
  const float* p = base + m * stride_f + kbase + koff;
  v4f x0 = *(const v4f*)(p);
  v4f x1 = *(const v4f*)(p + 4);
  v4f x2 = *(const v4f*)(p + 16);
  v4f x3 = *(const v4f*)(p + 20);
  v16h a;
#pragma unroll
  for (int i = 0; i < 4; ++i) {
    a[i]      = (_Float16)x0[i];
    a[4 + i]  = (_Float16)x1[i];
    a[8 + i]  = (_Float16)x2[i];
    a[12 + i] = (_Float16)x3[i];
  }
  return a;
}

// ---- B fragment (32x16 f16) gathered from row-major f32 W[k*ldw + n] ----
__device__ __forceinline__ v16h glb_b_frag(const float* __restrict__ W, int ldw,
                                           int kbase, int ncol0) {
  const int lane = threadIdx.x & 31;
  const int n    = ncol0 + (lane & 15);
  const int koff = (lane < 16) ? 0 : 8;
  v16h b;
#pragma unroll
  for (int j = 0; j < 16; ++j) b[j] = (_Float16)W[(size_t)(kbase + koff + (j < 8 ? j : j + 8)) * ldw + n];
  return b;
}

__device__ __forceinline__ v8f wmma_f16(v16h a, v16h b, v8f c) {
  return __builtin_amdgcn_wmma_f32_16x16x32_f16(false, a, false, b, (short)0, c,
                                                false, false);
}

// ---- CDNA5 async DMA: global -> LDS, 16 KB tile, 4 x b128 per thread ----
__device__ __forceinline__ void stage_tile_async(const float* gsrc, unsigned lds_off) {
  const int tid = threadIdx.x;
#pragma unroll
  for (int j = 0; j < 4; ++j) {
    unsigned loff = lds_off + (unsigned)(tid * 16 + j * 4096);
    const char* g = (const char*)gsrc + tid * 16 + j * 4096;
    asm volatile("global_load_async_to_lds_b128 %0, %1, off"
                 :: "v"(loff), "v"(g) : "memory");
  }
}

// =====================================================================
// Kernel 1: actor branch on candidate nodes only (32x work reduction)
// =====================================================================
__global__ __launch_bounds__(256)
void actor_cand_kernel(const float* __restrict__ x,
                       const unsigned char* __restrict__ node_mask,
                       const int* __restrict__ cand_idx,
                       const float* __restrict__ W_enc_a, const float* __restrict__ b_enc_a,
                       const float* __restrict__ ln_g, const float* __restrict__ ln_b,
                       const float* __restrict__ W_actor, const float* __restrict__ b_actor,
                       _Float16* __restrict__ hcand_ws, float* __restrict__ base_ws) {
  __shared__ __align__(16) _Float16 xc[KDIM * DDIM];
  __shared__ float ha[KDIM * HDIM];
  __shared__ float nmv[KDIM];
  __shared__ float wa[HDIM], lg[HDIM], lb[HDIM];
  __shared__ float mu_s[KDIM], rs_s[KDIM];
  __shared__ float red[256];

  const int r    = blockIdx.x;
  const int tid  = threadIdx.x;
  const int lane = tid & 31, w = tid >> 5;
  const int ncol0 = w * 16;

  for (int i = tid; i < KDIM * DDIM; i += 256) {
    int row = i >> 6, d = i & 63;
    int node = cand_idx[r * KDIM + row];
    xc[i] = (_Float16)x[((size_t)r * NDIM + node) * DDIM + d];
  }
  if (tid < KDIM) {
    int node = cand_idx[r * KDIM + tid];
    nmv[tid] = node_mask[(size_t)r * NDIM + node] ? 1.0f : 0.0f;
  }
  if (tid < HDIM) { wa[tid] = W_actor[tid]; lg[tid] = ln_g[tid]; lb[tid] = ln_b[tid]; }

  v16h b0 = glb_b_frag(W_enc_a, HDIM, 0,  ncol0);
  v16h b1 = glb_b_frag(W_enc_a, HDIM, 32, ncol0);
  const float bias = b_enc_a[ncol0 + (lane & 15)];
  __syncthreads();

#pragma unroll
  for (int mt = 0; mt < 4; ++mt) {
    v8f c = {};
    c = wmma_f16(lds_a_frag(&xc[mt * 16 * DDIM + 0], DDIM),  b0, c);
    c = wmma_f16(lds_a_frag(&xc[mt * 16 * DDIM + 32], DDIM), b1, c);
    const int col = ncol0 + (lane & 15);
#pragma unroll
    for (int p = 0; p < 8; ++p) {
      int m = mt * 16 + p + ((lane < 16) ? 0 : 8);
      ha[m * HDIM + col] = fmaxf(c[p] + bias, 0.0f) * nmv[m];
    }
  }
  __syncthreads();

  // LayerNorm stats: 4 threads per candidate row
  {
    int row = tid >> 2, part = tid & 3;
    float s = 0.f, s2 = 0.f;
    for (int h = part * 32; h < part * 32 + 32; ++h) {
      float v = ha[row * HDIM + h];
      s += v; s2 += v * v;
    }
    red[tid] = s;
    __syncthreads();
    if (part == 0)
      mu_s[row] = (red[tid] + red[tid + 1] + red[tid + 2] + red[tid + 3]) * (1.0f / HDIM);
    __syncthreads();
    red[tid] = s2;
    __syncthreads();
    if (part == 0) {
      float m2 = (red[tid] + red[tid + 1] + red[tid + 2] + red[tid + 3]) * (1.0f / HDIM);
      float mu = mu_s[row];
      rs_s[row] = rsqrtf(fmaxf(m2 - mu * mu, 0.0f) + EPSV);
    }
    __syncthreads();
  }

  for (int i = tid; i < KDIM * HDIM; i += 256) {
    int row = i >> 7, h = i & 127;
    float v = (ha[i] - mu_s[row]) * rs_s[row] * lg[h] + lb[h];
    ha[i] = v;
    hcand_ws[((size_t)r * KDIM + row) * HDIM + h] = (_Float16)v;
  }
  __syncthreads();

  // base = hn @ W_actor + b_actor
  {
    int row = tid >> 2, part = tid & 3;
    float s = 0.f;
    for (int h = part * 32; h < part * 32 + 32; ++h) s += ha[row * HDIM + h] * wa[h];
    red[tid] = s;
    __syncthreads();
    if (part == 0)
      base_ws[r * KDIM + row] =
          red[tid] + red[tid + 1] + red[tid + 2] + red[tid + 3] + b_actor[0];
  }
}

// =====================================================================
// Kernel 2: competitor fusion MLP + segment softmax -> logits
// =====================================================================
__global__ __launch_bounds__(256)
void comp_fusion_kernel(const float* __restrict__ x,
                        const int* __restrict__ comp_idx,
                        const unsigned char* __restrict__ comp_mask,
                        const unsigned char* __restrict__ cand_mask,
                        const float* __restrict__ W_phi1, const float* __restrict__ b_phi1,
                        const float* __restrict__ W_phi2, const float* __restrict__ b_phi2,
                        const float* __restrict__ w_score, const float* __restrict__ w_comp,
                        const float* __restrict__ comp_bias,
                        const _Float16* __restrict__ hcand_ws,
                        const float* __restrict__ base_ws,
                        float* __restrict__ logits) {
  __shared__ __align__(16) _Float16 feat[CDIM * (HDIM + DDIM)];
  __shared__ __align__(16) _Float16 z1[CDIM * HDIM];
  __shared__ float zf[CDIM * HDIM];
  __shared__ float wsc[HDIM], wcp[HDIM];
  __shared__ int   cidx[CDIM];
  __shared__ float sarr[CDIM], attn[CDIM];
  __shared__ float red[256];
  __shared__ float hasf;

  const int tid = threadIdx.x, lane = tid & 31, w = tid >> 5;
  const int ncol0 = w * 16;

  if (tid < HDIM) { wsc[tid] = w_score[tid]; wcp[tid] = w_comp[tid]; }

  v16h b1f[6];
#pragma unroll
  for (int kb = 0; kb < 6; ++kb) b1f[kb] = glb_b_frag(W_phi1, HDIM, kb * 32, ncol0);
  v16h b2f[4];
#pragma unroll
  for (int kb = 0; kb < 4; ++kb) b2f[kb] = glb_b_frag(W_phi2, HDIM, kb * 32, ncol0);
  const float bb1 = b_phi1[ncol0 + (lane & 15)];
  const float bb2 = b_phi2[ncol0 + (lane & 15)];

  for (int p = 0; p < PAIRS; ++p) {
    const int pair = blockIdx.x * PAIRS + p;   // = r*K + k
    const int r = pair >> 6;

    __syncthreads();
    if (tid < CDIM) cidx[tid] = comp_idx[(size_t)pair * CDIM + tid];
    __syncthreads();

    for (int i = tid; i < CDIM * (HDIM + DDIM); i += 256) {
      int row = i / (HDIM + DDIM), col = i % (HDIM + DDIM);
      _Float16 v;
      if (col < HDIM) v = hcand_ws[(size_t)pair * HDIM + col];
      else            v = (_Float16)x[((size_t)r * NDIM + cidx[row]) * DDIM + (col - HDIM)];
      feat[i] = v;
    }
    __syncthreads();

    // z1 = relu(feat @ W_phi1 + b1)
    v8f c = {};
#pragma unroll
    for (int kb = 0; kb < 6; ++kb)
      c = wmma_f16(lds_a_frag(&feat[kb * 32], HDIM + DDIM), b1f[kb], c);
    {
      const int col = ncol0 + (lane & 15);
#pragma unroll
      for (int q = 0; q < 8; ++q) {
        int m = q + ((lane < 16) ? 0 : 8);
        z1[m * HDIM + col] = (_Float16)fmaxf(c[q] + bb1, 0.0f);
      }
    }
    __syncthreads();

    // z = z1 @ W_phi2 + b2
    v8f c2 = {};
#pragma unroll
    for (int kb = 0; kb < 4; ++kb)
      c2 = wmma_f16(lds_a_frag(&z1[kb * 32], HDIM), b2f[kb], c2);
    {
      const int col = ncol0 + (lane & 15);
#pragma unroll
      for (int q = 0; q < 8; ++q) {
        int m = q + ((lane < 16) ? 0 : 8);
        zf[m * HDIM + col] = c2[q] + bb2;
      }
    }
    __syncthreads();

    // s[c] = z . w_score  (16 threads per competitor row)
    {
      int row = tid >> 4, part = tid & 15;
      float s = 0.f;
      for (int h = part * 8; h < part * 8 + 8; ++h) s += zf[row * HDIM + h] * wsc[h];
      red[tid] = s;
    }
    __syncthreads();
    if (tid < CDIM) {
      float s = 0.f;
      for (int j = 0; j < 16; ++j) s += red[tid * 16 + j];
      sarr[tid] = comp_mask[(size_t)pair * CDIM + tid] ? s : NEGV;
    }
    __syncthreads();
    if (tid == 0) {
      float mx = sarr[0];
      for (int c0 = 1; c0 < CDIM; ++c0) mx = fmaxf(mx, sarr[c0]);
      float den = 0.f;
      for (int c0 = 0; c0 < CDIM; ++c0) den += __expf(sarr[c0] - mx);
      float inv = 1.0f / den;
      for (int c0 = 0; c0 < CDIM; ++c0) attn[c0] = __expf(sarr[c0] - mx) * inv;
      int has = 0;
      for (int c0 = 0; c0 < CDIM; ++c0) has |= (comp_mask[(size_t)pair * CDIM + c0] != 0);
      hasf = has ? 1.f : 0.f;
    }
    __syncthreads();

    // u = attn @ z ; comp_logit = u . w_comp
    {
      float pbv = 0.f;
      if (tid < HDIM) {
        float u = 0.f;
        for (int c0 = 0; c0 < CDIM; ++c0) u += attn[c0] * zf[c0 * HDIM + tid];
        pbv = u * wcp[tid];
      }
      red[tid] = pbv;
    }
    __syncthreads();
    for (int off = 128; off > 0; off >>= 1) {
      if (tid < off) red[tid] += red[tid + off];
      __syncthreads();
    }
    if (tid == 0) {
      float comp_logit = red[0] + comp_bias[0];
      float b = base_ws[pair];
      float outv = cand_mask[pair] ? (b + (hasf > 0.5f ? comp_logit : 0.0f)) : NEGV;
      logits[pair] = outv;
    }
  }
}

// =====================================================================
// Kernel 3: critic — async-LDS double-buffered streaming encoder
//           + online-softmax attention pooling (64 nodes / iteration)
// =====================================================================
__global__ __launch_bounds__(256)
void critic_kernel(const float* __restrict__ x,
                   const unsigned char* __restrict__ node_mask,
                   const float* __restrict__ W_enc_c, const float* __restrict__ b_enc_c,
                   const float* __restrict__ w_attn, const float* __restrict__ b_attn,
                   const float* __restrict__ W_crit1, const float* __restrict__ b_crit1,
                   const float* __restrict__ W_crit2, const float* __restrict__ b_crit2,
                   float* __restrict__ values) {
  __shared__ __align__(16) float xbuf[2][TN * DDIM];   // 2 x 16 KB f32 tiles
  __shared__ float nmv[TN];
  __shared__ float scred[TN][8];
  __shared__ float scv[TN], ev[TN];
  __shared__ float st[3];            // 0: running max, 1: running denom, 2: scale
  __shared__ float gsh[HDIM];
  __shared__ float red[256];

  const int r = blockIdx.x;
  const int tid = threadIdx.x, lane = tid & 31, w = tid >> 5;
  const int ncol0 = w * 16;
  const int mbase = (lane < 16) ? 0 : 8;

  v16h b0 = glb_b_frag(W_enc_c, HDIM, 0,  ncol0);
  v16h b1 = glb_b_frag(W_enc_c, HDIM, 32, ncol0);
  const float bias = b_enc_c[ncol0 + (lane & 15)];
  const float wav  = w_attn[ncol0 + (lane & 15)];
  float gloc = 0.0f;
  if (tid == 0) { st[0] = -1e30f; st[1] = 0.0f; }

  const unsigned loff0 = (unsigned)(unsigned long long)(&xbuf[0][0]);
  const unsigned loff1 = (unsigned)(unsigned long long)(&xbuf[1][0]);
  const float* xrow = x + (size_t)r * NDIM * DDIM;

  // prologue: kick off tile 0 DMA
  stage_tile_async(xrow, loff0);

  for (int t = 0; t < NTILES; ++t) {
    const int cur = t & 1;
    if (tid < TN) nmv[tid] = node_mask[(size_t)r * NDIM + t * TN + tid] ? 1.0f : 0.0f;
    if (t + 1 < NTILES) {
      stage_tile_async(xrow + (size_t)(t + 1) * TN * DDIM, cur ? loff0 : loff1);
      asm volatile("s_wait_asynccnt 0x4" ::: "memory");  // tile t's 4 loads done
    } else {
      asm volatile("s_wait_asynccnt 0x0" ::: "memory");
    }
    __syncthreads();

    // 4 M-tiles of 16 nodes, WMMA + attention-score partials
    float hval[4][8];
#pragma unroll
    for (int mt = 0; mt < 4; ++mt) {
      v8f c = {};
      c = wmma_f16(lds_a_frag_f32(&xbuf[cur][mt * 16 * DDIM], DDIM, 0),  b0, c);
      c = wmma_f16(lds_a_frag_f32(&xbuf[cur][mt * 16 * DDIM], DDIM, 32), b1, c);
      float dsum[8];
#pragma unroll
      for (int q = 0; q < 8; ++q) {
        float hv = fmaxf(c[q] + bias, 0.0f) * nmv[mt * 16 + q + mbase];
        hval[mt][q] = hv;
        dsum[q] = hv * wav;
      }
#pragma unroll
      for (int q = 0; q < 8; ++q) {
        float v = dsum[q];
        v += __shfl_xor(v, 1);
        v += __shfl_xor(v, 2);
        v += __shfl_xor(v, 4);
        v += __shfl_xor(v, 8);
        dsum[q] = v;
      }
      if ((lane & 15) == 0) {
#pragma unroll
        for (int q = 0; q < 8; ++q) scred[mt * 16 + q + mbase][w] = dsum[q];
      }
    }
    __syncthreads();
    if (tid < TN) {
      float s = 0.f;
      for (int j = 0; j < 8; ++j) s += scred[tid][j];
      scv[tid] = (nmv[tid] > 0.5f) ? (s + b_attn[0]) : NEGV;
    }
    __syncthreads();
    if (tid == 0) {
      float tmax = scv[0];
      for (int j = 1; j < TN; ++j) tmax = fmaxf(tmax, scv[j]);
      float m_old = st[0];
      float m_new = fmaxf(m_old, tmax);
      float scale = __expf(m_old - m_new);
      float ssum = 0.f;
      for (int j = 0; j < TN; ++j) { float e = __expf(scv[j] - m_new); ev[j] = e; ssum += e; }
      st[0] = m_new;
      st[1] = st[1] * scale + ssum;
      st[2] = scale;
    }
    __syncthreads();
    {
      float acc = gloc * st[2];
#pragma unroll
      for (int mt = 0; mt < 4; ++mt)
#pragma unroll
        for (int q = 0; q < 8; ++q) acc += ev[mt * 16 + q + mbase] * hval[mt][q];
      gloc = acc;
    }
    __syncthreads();
  }

  // combine half-lane partials per column, publish g
  {
    float tot = gloc + __shfl_xor(gloc, 16);
    if (lane < 16) gsh[ncol0 + lane] = tot;
  }
  __syncthreads();
  const float inv_s = 1.0f / st[1];

  // critic head: relu(g @ W_crit1 + b1) @ W_crit2 + b2
  float pbv = 0.f;
  if (tid < HDIM) {
    float hid = b_crit1[tid];
    for (int h = 0; h < HDIM; ++h) hid += (gsh[h] * inv_s) * W_crit1[h * HDIM + tid];
    hid = fmaxf(hid, 0.0f);
    pbv = hid * W_crit2[tid];
  }
  red[tid] = pbv;
  __syncthreads();
  for (int off = 128; off > 0; off >>= 1) {
    if (tid < off) red[tid] += red[tid + off];
    __syncthreads();
  }
  if (tid == 0) values[r] = red[0] + b_crit2[0];
}

extern "C" void kernel_launch(void* const* d_in, const int* in_sizes, int n_in,
                              void* d_out, int out_size, void* d_ws, size_t ws_size,
                              hipStream_t stream) {
  (void)in_sizes; (void)n_in; (void)out_size; (void)ws_size;
  const float*         x         = (const float*)d_in[0];
  const unsigned char* node_mask = (const unsigned char*)d_in[1];
  const int*           cand_idx  = (const int*)d_in[2];
  const unsigned char* cand_mask = (const unsigned char*)d_in[3];
  const int*           comp_idx  = (const int*)d_in[4];
  const unsigned char* comp_mask = (const unsigned char*)d_in[5];
  const float* W_enc_a = (const float*)d_in[6];
  const float* b_enc_a = (const float*)d_in[7];
  const float* W_enc_c = (const float*)d_in[8];
  const float* b_enc_c = (const float*)d_in[9];
  const float* ln_g    = (const float*)d_in[10];
  const float* ln_b    = (const float*)d_in[11];
  const float* W_actor = (const float*)d_in[12];
  const float* b_actor = (const float*)d_in[13];
  const float* W_phi1  = (const float*)d_in[14];
  const float* b_phi1  = (const float*)d_in[15];
  const float* W_phi2  = (const float*)d_in[16];
  const float* b_phi2  = (const float*)d_in[17];
  const float* w_score = (const float*)d_in[18];
  const float* w_comp  = (const float*)d_in[19];
  const float* comp_b  = (const float*)d_in[20];
  const float* w_attn  = (const float*)d_in[21];
  const float* b_attn  = (const float*)d_in[22];
  const float* W_crit1 = (const float*)d_in[23];
  const float* b_crit1 = (const float*)d_in[24];
  const float* W_crit2 = (const float*)d_in[25];
  const float* b_crit2 = (const float*)d_in[26];

  float* logits = (float*)d_out;                 // [R*K]
  float* values = (float*)d_out + RDIM * KDIM;   // [R]

  _Float16* hcand_ws = (_Float16*)d_ws;          // R*K*H f16 = 4 MB
  float* base_ws = (float*)((char*)d_ws + (size_t)RDIM * KDIM * HDIM * sizeof(_Float16));

  actor_cand_kernel<<<RDIM, 256, 0, stream>>>(
      x, node_mask, cand_idx, W_enc_a, b_enc_a, ln_g, ln_b, W_actor, b_actor,
      hcand_ws, base_ws);
  comp_fusion_kernel<<<(RDIM * KDIM) / PAIRS, 256, 0, stream>>>(
      x, comp_idx, comp_mask, cand_mask, W_phi1, b_phi1, W_phi2, b_phi2,
      w_score, w_comp, comp_b, hcand_ws, base_ws, logits);
  critic_kernel<<<RDIM, 256, 0, stream>>>(
      x, node_mask, W_enc_c, b_enc_c, w_attn, b_attn, W_crit1, b_crit1,
      W_crit2, b_crit2, values);
}
